// LSTM_54537494725306
// MI455X (gfx1250) — matrix operand order repaired and checked by reading, not compiled, not run on hardware
//
#include <hip/hip_runtime.h>
#include <hip/hip_bf16.h>

#define T_STEPS 2048
#define DIM_D   512
#define DIM_H   2048
#define DIM_G   (4*DIM_H)   // 8192
#define DIM_OUT 64
#define NWG     128
#define ROWS_PER_WG   (DIM_G / NWG)   // 64 rows -> 4 tiles of 16
#define KSPLIT        2
#define KS            (DIM_H / KSPLIT) // 1024 per K-slice

typedef __attribute__((ext_vector_type(16))) __bf16 v16bf;
typedef __attribute__((ext_vector_type(8)))  __bf16 v8bf;
typedef __attribute__((ext_vector_type(8)))  float  v8f;
typedef __attribute__((ext_vector_type(4)))  int    v4i;
typedef __attribute__((address_space(1))) v4i v4i_g;   // global int4
typedef __attribute__((address_space(3))) v4i v4i_l;   // LDS int4

union ABu { v16bf v; v8bf h[2]; };

__device__ __forceinline__ float fsig(float x){ return 1.0f/(1.0f+__expf(-x)); }

__device__ __forceinline__ v8f wmma_bf16(v16bf a, v16bf b, v8f c){
  // (neg_a, A, neg_b, B, c_mod, C, reuse_a, reuse_b)
  return __builtin_amdgcn_wmma_f32_16x16x32_bf16(false, a, false, b, (short)0, c, false, false);
}

// One wave: 16-row GEMV tile over K elements. wrow pre-offset to
// W + (row0 + (lane&15))*ld + kbase. h broadcast into all 16 WMMA columns.
__device__ __forceinline__ v8f gemv_tile(const __bf16* __restrict__ wrow,
                                         const __bf16* hl, int K, v8f acc, int lane){
  const int sa = (lane < 16) ? 0 : 8;    // 16-bit A layout halves
  const int sb = (lane < 16) ? 0 : 8;    // 16-bit B layout halves (same as A)
  for (int k0 = 0; k0 < K; k0 += 32){
    ABu A, B;
    A.h[0] = *(const v8bf*)(wrow + k0 + sa);
    A.h[1] = *(const v8bf*)(wrow + k0 + 16 + sa);
    B.h[0] = *(const v8bf*)(hl + k0 + sb);
    B.h[1] = *(const v8bf*)(hl + k0 + 16 + sb);
    if (k0 + 256 < K) __builtin_prefetch(wrow + k0 + 256, 0, 0);  // global_prefetch_b8
    acc = wmma_bf16(A.v, B.v, acc);
  }
  return acc;
}

// Combine KSPLIT partial accumulators through LDS; emit(final_local_row, value).
template <typename EmitF>
__device__ __forceinline__ void reduce_emit(float (*red)[16], const v8f& acc,
                                            int wid, int lane, EmitF emit){
  if (lane == 0){
    #pragma unroll
    for (int r = 0; r < 8; ++r) red[wid][r] = acc[r];
  } else if (lane == 16){
    #pragma unroll
    for (int r = 0; r < 8; ++r) red[wid][8+r] = acc[r];
  }
  __syncthreads();
  if ((wid & 1) == 0 && lane < 16){
    float s = red[wid][lane] + red[wid+1][lane];
    emit((wid >> 1)*16 + lane, s);
  }
  __syncthreads();
}

__device__ __forceinline__ void gbar(unsigned* bar, unsigned target){
  __threadfence();
  __syncthreads();
  if (threadIdx.x == 0){
    __hip_atomic_fetch_add(bar, 1u, __ATOMIC_ACQ_REL, __HIP_MEMORY_SCOPE_AGENT);
    while (__hip_atomic_load(bar, __ATOMIC_ACQUIRE, __HIP_MEMORY_SCOPE_AGENT) < target)
      __builtin_amdgcn_s_sleep(1);
  }
  __syncthreads();
}

// ---------------------------------------------------------------- prep: casts
__global__ void k_prep(const float* __restrict__ Wih1, const float* __restrict__ Whh1,
                       const float* __restrict__ Wih2, const float* __restrict__ Whh2,
                       const float* __restrict__ Wlin, const float* __restrict__ x,
                       const float* __restrict__ bih1, const float* __restrict__ bhh1,
                       const float* __restrict__ bih2, const float* __restrict__ bhh2,
                       __bf16* wih1b, __bf16* whh1b, __bf16* wih2b, __bf16* whh2b,
                       __bf16* wlinb, __bf16* xb, float* bias1, float* bias2,
                       float* c1buf, float* c2buf, unsigned* bar)
{
  const size_t NW1 = (size_t)DIM_G*DIM_D;
  const size_t NWH = (size_t)DIM_G*DIM_H;
  const size_t NWL = (size_t)DIM_OUT*DIM_H;
  const size_t NX  = (size_t)T_STEPS*DIM_D;
  const size_t total = NW1 + 3*NWH + NWL + NX + 2*(size_t)DIM_G + 4*(size_t)DIM_H + 1;
  for (size_t i = blockIdx.x*(size_t)blockDim.x + threadIdx.x; i < total;
       i += (size_t)gridDim.x*blockDim.x){
    size_t j = i;
    if (j < NW1){ wih1b[j] = (__bf16)Wih1[j]; continue; } j -= NW1;
    if (j < NWH){ whh1b[j] = (__bf16)Whh1[j]; continue; } j -= NWH;
    if (j < NWH){ wih2b[j] = (__bf16)Wih2[j]; continue; } j -= NWH;
    if (j < NWH){ whh2b[j] = (__bf16)Whh2[j]; continue; } j -= NWH;
    if (j < NWL){ wlinb[j] = (__bf16)Wlin[j]; continue; } j -= NWL;
    if (j < NX ){ xb[j]    = (__bf16)x[j];    continue; } j -= NX;
    if (j < DIM_G){ bias1[j] = bih1[j] + bhh1[j]; continue; } j -= DIM_G;
    if (j < DIM_G){ bias2[j] = bih2[j] + bhh2[j]; continue; } j -= DIM_G;
    if (j < 2*DIM_H){ c1buf[j] = 0.0f; continue; } j -= 2*DIM_H;
    if (j < 2*DIM_H){ c2buf[j] = 0.0f; continue; } j -= 2*DIM_H;
    *bar = 0u;
  }
}

// --------------------------- phase 1: X1 = x @ W_ih1^T + bias1 (dense WMMA GEMM)
__global__ void __launch_bounds__(256) k_gemm_x(const __bf16* __restrict__ xb,
                                                const __bf16* __restrict__ wih1b,
                                                const float* __restrict__ bias1,
                                                float* __restrict__ X1)
{
  const int wid  = threadIdx.x >> 5;
  const int lane = threadIdx.x & 31;
  const int tile = blockIdx.x*8 + wid;        // 65536 = 128(M) x 512(N) tiles
  const int mt = tile & 127, nt = tile >> 7;
  const int m0 = mt*16, n0 = nt*16;
  const __bf16* ap = xb    + (size_t)(m0 + (lane & 15))*DIM_D;
  const __bf16* bp = wih1b + (size_t)(n0 + (lane & 15))*DIM_D;
  const int sa = (lane < 16) ? 0 : 8;
  const int sb = (lane < 16) ? 0 : 8;
  v8f acc = {};
  for (int k0 = 0; k0 < DIM_D; k0 += 32){
    ABu A, B;
    A.h[0] = *(const v8bf*)(ap + k0 + sa);
    A.h[1] = *(const v8bf*)(ap + k0 + 16 + sa);
    B.h[0] = *(const v8bf*)(bp + k0 + sb);
    B.h[1] = *(const v8bf*)(bp + k0 + 16 + sb);
    acc = wmma_bf16(A.v, B.v, acc);
  }
  const int n = n0 + (lane & 15);
  const float b = bias1[n];
  const int mbase = m0 + ((lane < 16) ? 0 : 8);
  for (int r = 0; r < 8; ++r)
    X1[(size_t)(mbase + r)*DIM_G + n] = acc[r] + b;
}

// ------------------------------ phase 2: persistent recurrent kernel
__global__ void __launch_bounds__(256, 1) k_lstm_persistent(
    const __bf16* __restrict__ whh1b, const __bf16* __restrict__ wih2b,
    const __bf16* __restrict__ whh2b, const __bf16* __restrict__ wlinb,
    const float* __restrict__ X1, const float* __restrict__ bias2,
    const float* __restrict__ blin,
    float* c1buf, float* c2buf, float* g1, float* g2,
    unsigned* bar, float* __restrict__ yout)
{
  // 320KB WGP LDS: 256KB layer-1 weight slice + h-state + reduction pad
  __shared__ __align__(16) __bf16 wslice[ROWS_PER_WG * DIM_H];  // 256 KB
  __shared__ __align__(16) __bf16 h1s[DIM_H];                   // 4 KB
  __shared__ __align__(16) __bf16 h2s[DIM_H];                   // 4 KB
  __shared__ float red[8][16];                                  // 512 B
  const int wg   = blockIdx.x;
  const int tid  = threadIdx.x;
  const int wid  = tid >> 5;
  const int lane = tid & 31;
  const int tile = wid >> 1;        // 0..3 (16-row tile within WG slice)
  const int ks   = wid & 1;         // K-slice 0..1
  const int row0 = wg*ROWS_PER_WG;  // first gate row owned by this WG
  unsigned bt = 0;

  // ---- stage layer-1 recurrent weight slice into LDS (async DMA if available)
  {
    const size_t SLICE_BYTES = (size_t)ROWS_PER_WG * DIM_H * sizeof(__bf16);
    const char* gsl = (const char*)(whh1b + (size_t)row0 * DIM_H);
    char* lsl = (char*)wslice;
    for (size_t off = (size_t)tid * 16; off < SLICE_BYTES; off += 256u * 16u){
#if __has_builtin(__builtin_amdgcn_global_load_async_to_lds_b128)
      __builtin_amdgcn_global_load_async_to_lds_b128(
          (v4i_g*)(gsl + off), (v4i_l*)(lsl + off), 0, 0);
#else
      *(v8bf*)(lsl + off) = *(const v8bf*)(gsl + off);
#endif
    }
#if __has_builtin(__builtin_amdgcn_global_load_async_to_lds_b128)
#if __has_builtin(__builtin_amdgcn_s_wait_asynccnt)
    __builtin_amdgcn_s_wait_asynccnt(0);
#else
    asm volatile("s_wait_asynccnt 0x0" ::: "memory");
#endif
#endif
  }
  __syncthreads();

  for (int t = 0; t < T_STEPS; ++t){
    // -- phase 1: reconstruct h2(t-1) into LDS; publish c2(t-1); y[t-1] on WG0
    if (t == 0){
      for (int j = tid; j < DIM_H; j += 256){ h1s[j] = (__bf16)0.0f; h2s[j] = (__bf16)0.0f; }
    } else {
      const float* g2p = g2 + ((t-1) & 1)*DIM_G;
      const float* c2r = c2buf + (t & 1)*DIM_H;        // c2(t-2)
      float*       c2w = c2buf + ((t-1) & 1)*DIM_H;    // c2(t-1) (all WGs write same value)
      for (int j = tid; j < DIM_H; j += 256){
        float i2 = fsig(g2p[j]);
        float f2 = fsig(g2p[DIM_H + j]);
        float gg = tanhf(g2p[2*DIM_H + j]);
        float o2 = fsig(g2p[3*DIM_H + j]);
        float c2n = f2*c2r[j] + i2*gg;
        c2w[j] = c2n;
        h2s[j] = (__bf16)(o2*tanhf(c2n));
      }
    }
    __syncthreads();

    if (t > 0 && wg == 0){                 // y[t-1] = W_lin @ h2(t-1) + b_lin
      v8f acc = {};
      const __bf16* wr = wlinb + (size_t)(tile*16 + (lane & 15))*DIM_H + ks*KS;
      acc = gemv_tile(wr, h2s + ks*KS, KS, acc, lane);
      float* yrow = yout + (size_t)(t-1)*DIM_OUT;
      const float* bl = blin;
      reduce_emit(red, acc, wid, lane, [&](int lr, float s){ yrow[lr] = s + bl[lr]; });
    }

    // gates1(t) = X1[t] + W_hh1 @ h1(t-1)   (A-operand from LDS weight cache)
    {
      v8f acc = {};
      const __bf16* wr = wslice + (size_t)(tile*16 + (lane & 15))*DIM_H + ks*KS;
      acc = gemv_tile(wr, h1s + ks*KS, KS, acc, lane);
      const float* x1r = X1 + (size_t)t*DIM_G;
      float* g1w = g1 + (t & 1)*DIM_G;
      reduce_emit(red, acc, wid, lane, [&](int lr, float s){
        g1w[row0 + lr] = s + x1r[row0 + lr];
      });
    }
    bt += NWG; gbar(bar, bt);                    // B1

    // -- phase 2: h1(t), c1(t); then gates2(t)
    {
      const float* g1p = g1 + (t & 1)*DIM_G;
      const float* c1r = c1buf + ((t-1) & 1)*DIM_H;
      float*       c1w = c1buf + (t & 1)*DIM_H;
      for (int j = tid; j < DIM_H; j += 256){
        float i1 = fsig(g1p[j]);
        float f1 = fsig(g1p[DIM_H + j]);
        float gg = tanhf(g1p[2*DIM_H + j]);
        float o1 = fsig(g1p[3*DIM_H + j]);
        float c1n = f1*c1r[j] + i1*gg;
        c1w[j] = c1n;
        h1s[j] = (__bf16)(o1*tanhf(c1n));
      }
      __syncthreads();
      // gates2(t) = bias2 + W_ih2 @ h1(t) + W_hh2 @ h2(t-1)
      v8f acc = {};
      const __bf16* wr1 = wih2b + (size_t)(row0 + tile*16 + (lane & 15))*DIM_H + ks*KS;
      acc = gemv_tile(wr1, h1s + ks*KS, KS, acc, lane);
      const __bf16* wr2 = whh2b + (size_t)(row0 + tile*16 + (lane & 15))*DIM_H + ks*KS;
      acc = gemv_tile(wr2, h2s + ks*KS, KS, acc, lane);
      float* g2w = g2 + (t & 1)*DIM_G;
      reduce_emit(red, acc, wid, lane, [&](int lr, float s){
        g2w[row0 + lr] = s + bias2[row0 + lr];
      });
    }
    bt += NWG; gbar(bar, bt);                    // B2
  }

  // epilogue: y[T-1]
  if (wg == 0){
    const float* g2p = g2 + ((T_STEPS-1) & 1)*DIM_G;
    const float* c2r = c2buf + (T_STEPS & 1)*DIM_H;  // c2(T-2)
    for (int j = tid; j < DIM_H; j += 256){
      float i2 = fsig(g2p[j]);
      float f2 = fsig(g2p[DIM_H + j]);
      float gg = tanhf(g2p[2*DIM_H + j]);
      float o2 = fsig(g2p[3*DIM_H + j]);
      float c2n = f2*c2r[j] + i2*gg;
      h2s[j] = (__bf16)(o2*tanhf(c2n));
    }
    __syncthreads();
    v8f acc = {};
    const __bf16* wr = wlinb + (size_t)(tile*16 + (lane & 15))*DIM_H + ks*KS;
    acc = gemv_tile(wr, h2s + ks*KS, KS, acc, lane);
    float* yrow = yout + (size_t)(T_STEPS-1)*DIM_OUT;
    reduce_emit(red, acc, wid, lane, [&](int lr, float s){ yrow[lr] = s + blin[lr]; });
  }
}

extern "C" void kernel_launch(void* const* d_in, const int* in_sizes, int n_in,
                              void* d_out, int out_size, void* d_ws, size_t ws_size,
                              hipStream_t stream)
{
  const float* x    = (const float*)d_in[0];
  const float* Wih1 = (const float*)d_in[1];
  const float* Whh1 = (const float*)d_in[2];
  const float* bih1 = (const float*)d_in[3];
  const float* bhh1 = (const float*)d_in[4];
  const float* Wih2 = (const float*)d_in[5];
  const float* Whh2 = (const float*)d_in[6];
  const float* bih2 = (const float*)d_in[7];
  const float* bhh2 = (const float*)d_in[8];
  const float* Wlin = (const float*)d_in[9];
  const float* blin = (const float*)d_in[10];
  float* yout = (float*)d_out;
  (void)in_sizes; (void)n_in; (void)out_size; (void)ws_size;

  char* base = (char*)d_ws;
  size_t off = 0;
  auto alloc = [&](size_t bytes) -> char* {
    off = (off + 255) & ~(size_t)255;
    char* p = base + off; off += bytes; return p;
  };
  __bf16* whh1b = (__bf16*)alloc((size_t)DIM_G*DIM_H*2);
  __bf16* wih2b = (__bf16*)alloc((size_t)DIM_G*DIM_H*2);
  __bf16* whh2b = (__bf16*)alloc((size_t)DIM_G*DIM_H*2);
  __bf16* wih1b = (__bf16*)alloc((size_t)DIM_G*DIM_D*2);
  __bf16* wlinb = (__bf16*)alloc((size_t)DIM_OUT*DIM_H*2);
  __bf16* xb    = (__bf16*)alloc((size_t)T_STEPS*DIM_D*2);
  float* X1     = (float*)alloc((size_t)T_STEPS*DIM_G*4);
  float* bias1  = (float*)alloc((size_t)DIM_G*4);
  float* bias2  = (float*)alloc((size_t)DIM_G*4);
  float* c1buf  = (float*)alloc((size_t)2*DIM_H*4);
  float* c2buf  = (float*)alloc((size_t)2*DIM_H*4);
  float* g1     = (float*)alloc((size_t)2*DIM_G*4);
  float* g2     = (float*)alloc((size_t)2*DIM_G*4);
  unsigned* bar = (unsigned*)alloc(256);

  hipLaunchKernelGGL(k_prep, dim3(4096), dim3(256), 0, stream,
                     Wih1, Whh1, Wih2, Whh2, Wlin, x, bih1, bhh1, bih2, bhh2,
                     wih1b, whh1b, wih2b, whh2b, wlinb, xb, bias1, bias2,
                     c1buf, c2buf, bar);
  hipLaunchKernelGGL(k_gemm_x, dim3(8192), dim3(256), 0, stream,
                     xb, wih1b, bias1, X1);
  hipLaunchKernelGGL(k_lstm_persistent, dim3(NWG), dim3(256), 0, stream,
                     whh1b, wih2b, whh2b, wlinb, X1, bias2, blin,
                     c1buf, c2buf, g1, g2, bar, yout);
}
